// Multiheadselfattention_45079976738981
// MI455X (gfx1250) — matrix operand order repaired and checked by reading, not compiled, not run on hardware
//
#include <hip/hip_runtime.h>

typedef __attribute__((ext_vector_type(16))) _Float16 v16h;
typedef __attribute__((ext_vector_type(8)))  _Float16 v8h;
typedef __attribute__((ext_vector_type(8)))  float    v8f;
typedef __attribute__((ext_vector_type(4)))  unsigned int v4u;
typedef __attribute__((ext_vector_type(8)))  int v8i;
typedef __attribute__((ext_vector_type(4)))  int v4i;

// D = A*B + C, A/B f16 16x32 / 32x16, C/D f32 16x16
#define WMMA_F16(a, b, c) \
  __builtin_amdgcn_wmma_f32_16x16x32_f16(false, (a), false, (b), (short)0, (c), false, false)

constexpr int BQ = 2;           // batch
constexpr int SEQ = 2048;       // sequence
constexpr int DM = 1024;        // d_model
constexpr int NH = 16;          // heads
constexpr int HD = 64;          // head dim
constexpr int MROWS = BQ * SEQ; // 4096

__device__ __forceinline__ v16h cat8(v8h lo, v8h hi) {
  return __builtin_shufflevector(lo, hi, 0,1,2,3,4,5,6,7,8,9,10,11,12,13,14,15);
}

// A-operand fragment (16x32, row per lane, contiguous 16 K values per lane half).
__device__ __forceinline__ v16h load_a_frag(const _Float16* row, int kbase, int half) {
  return *(const v16h*)(row + kbase + half * 16);
}

// B-operand fragment (32x16): lane = column, contiguous 16 K values per lane half.
// Source is stored as N rows of contiguous K, so this is one 32B load per lane.
__device__ __forceinline__ v16h load_b_frag(const _Float16* row, int kbase, int half) {
  return *(const v16h*)(row + kbase + half * 16);
}

// ---------------------------------------------------------- TDM tile load ---
// DMA a 64-row x 32-element (f16) tile of a [1024 x 1024] f16 matrix
// (row stride 1024 elements) into LDS, packed row-major [64][32].
// D# bit layout per CDNA5 ISA 8.3/8.4; tracked by TENSORcnt.
__device__ __forceinline__ void tdm_load_b_tile(const _Float16* gsrc, unsigned lds_off) {
  const unsigned long long ga = (unsigned long long)(size_t)gsrc;
  v4u g0 = { 1u,                               // count=1, user descriptor
             lds_off,                          // LDS byte address
             (unsigned)ga,                     // global_addr[31:0]
             ((unsigned)(ga >> 32) & 0x01FFFFFFu) | 0x80000000u }; // [56:32] | type=2
  v8i g1 = { 0x00010000,    // workgroup_mask=0, data_size=1 (2 bytes)
             0x04000000,    // tensor_dim0[15:0]=1024 in [31:16]
             0x04000000,    // tensor_dim0 hi=0, tensor_dim1[15:0]=1024 in [31:16]
             0x00200000,    // tensor_dim1 hi=0, tile_dim0=32 in [31:16]
             64,            // tile_dim1=64, tile_dim2=0
             1024,          // tensor_dim0_stride lo = 1024 elements
             0, 0 };
  v4i z = { 0, 0, 0, 0 };
#if __clang_major__ >= 23
  v8i z8 = { 0, 0, 0, 0, 0, 0, 0, 0 };
  __builtin_amdgcn_tensor_load_to_lds(g0, g1, z, z, z8, 0);
#else
  __builtin_amdgcn_tensor_load_to_lds(g0, g1, z, z, 0);
#endif
}

// ---------------------------------------------------------------- convert ---
__global__ __launch_bounds__(256) void f32_to_f16_k(const float* __restrict__ in,
                                                    _Float16* __restrict__ out, int n) {
  int i = blockIdx.x * 256 + threadIdx.x;
  if (i < n) out[i] = (_Float16)in[i];
}

// ------------------------------------------------------------------- GEMM ---
// C[M,N] = A[M,K] * W[N,K]^T  with M=4096, N=K=1024.
// The 64x32 f16 B tile is DMA'd into LDS by the Tensor Data Mover (wave 0),
// double-buffered: while WMMA consumes buffer b, the TDM fills buffer b^1.
// A fragments are wave-private -> direct global loads.
// LAYOUT 0: f16 out as [B,H,S,hd]   (q, k)
// LAYOUT 1: f16 out as [B,H,hd,S]   (v transposed)
// LAYOUT 2: f32 out as [M,N]        (final projection)
template <int LAYOUT>
__global__ __launch_bounds__(256) void gemm_wmma_k(const _Float16* __restrict__ A,
                                                   const _Float16* __restrict__ W,
                                                   void* __restrict__ Cout) {
  alignas(128) __shared__ _Float16 lds_b[2][64 * 32];   // two 4KB tiles
  const int tid  = threadIdx.x;
  const int lane = tid & 31;
  const int wv   = tid >> 5;   // 0..7
  const int l15  = lane & 15;
  const int half = lane >> 4;
  const int m0 = blockIdx.x * 128 + wv * 16;
  const int n0 = blockIdx.y * 64;

  v8f acc[4] = {};
  const _Float16* arow  = A + (size_t)(m0 + l15) * DM;
  const _Float16* wbase = W + (size_t)n0 * DM;

  if (wv == 0)
    tdm_load_b_tile(wbase, (unsigned)(size_t)&lds_b[0][0]);

  for (int kk = 0; kk < DM; kk += 32) {
    const int buf = (kk >> 5) & 1;
    if (wv == 0)
      __builtin_amdgcn_s_wait_tensorcnt(0);   // tile `buf` landed in LDS
    __syncthreads();                          // visible to all 8 waves
    if (wv == 0 && kk + 32 < DM)              // DMA next tile during WMMA
      tdm_load_b_tile(wbase + kk + 32, (unsigned)(size_t)&lds_b[buf ^ 1][0]);

    v16h a = load_a_frag(arow, kk, half);
#pragma unroll
    for (int t = 0; t < 4; ++t) {
      v16h b = *(const v16h*)(&lds_b[buf][(t * 16 + l15) * 32 + half * 16]);
      acc[t] = WMMA_F16(a, b, acc[t]);
    }
  }

#pragma unroll
  for (int t = 0; t < 4; ++t) {
    const int n = n0 + t * 16 + l15;
#pragma unroll
    for (int r = 0; r < 8; ++r) {
      const int m = m0 + half * 8 + r;
      const float v = acc[t][r];
      if (LAYOUT == 2) {
        ((float*)Cout)[(size_t)m * DM + n] = v;
      } else {
        const int b = m >> 11, s = m & (SEQ - 1);
        const int h = n >> 6,  d = n & (HD - 1);
        if (LAYOUT == 0)
          ((_Float16*)Cout)[(((size_t)(b * NH + h) * SEQ) + s) * HD + d] = (_Float16)v;
        else
          ((_Float16*)Cout)[(((size_t)(b * NH + h) * HD) + d) * SEQ + s] = (_Float16)v;
      }
    }
  }
}

// ------------------------------------------------------------------- RoPE ---
__global__ __launch_bounds__(256) void rope_k(_Float16* __restrict__ q,
                                              _Float16* __restrict__ k) {
  const int idx  = blockIdx.x * 256 + threadIdx.x;  // over B*H*S*32
  const int pair = idx & 31;
  const int s    = (idx >> 5) & (SEQ - 1);
  const int bh   = idx >> 16;
  const float theta = __powf(10000.0f, -(float)(2 * pair) / 64.0f);
  const float ang = (float)s * theta;
  const float c = __cosf(ang), sn = __sinf(ang);
  const size_t base = (((size_t)bh * SEQ) + s) * HD + 2 * pair;
  const float q1 = (float)q[base], q2 = (float)q[base + 1];
  q[base]     = (_Float16)(q1 * c - q2 * sn);
  q[base + 1] = (_Float16)(q1 * sn + q2 * c);
  const float k1 = (float)k[base], k2 = (float)k[base + 1];
  k[base]     = (_Float16)(k1 * c - k2 * sn);
  k[base + 1] = (_Float16)(k1 * sn + k2 * c);
}

// ---------------------------------------------------- flash attention core ---
// One 32-key step of online-softmax attention for a 16-row query tile.
// MASK=true only for the single tile that straddles the causal diagonal.
template <bool MASK>
__device__ __forceinline__ void attn_step(const _Float16* __restrict__ Kp,
                                          const _Float16* __restrict__ Vp,
                                          _Float16* ldsp,
                                          const v16h& qa0, const v16h& qa1,
                                          v8f (&acc)[4], float (&mrow)[8],
                                          float (&lrow)[8],
                                          int j, int q0, int l15, int half) {
  // ---- scores: two 16x16 tiles covering keys [j, j+32)
  v8f s0 = {}, s1 = {};
  {
    v16h kb = load_b_frag(Kp + (size_t)(j + l15) * HD, 0, half);
    s0 = WMMA_F16(qa0, kb, s0);
    kb = load_b_frag(Kp + (size_t)(j + l15) * HD, 32, half);
    s0 = WMMA_F16(qa1, kb, s0);
    kb = load_b_frag(Kp + (size_t)(j + 16 + l15) * HD, 0, half);
    s1 = WMMA_F16(qa0, kb, s1);
    kb = load_b_frag(Kp + (size_t)(j + 16 + l15) * HD, 32, half);
    s1 = WMMA_F16(qa1, kb, s1);
  }
  if (!MASK)  // warm the next K tile near the WGP (global_prefetch_b8)
    __builtin_prefetch(Kp + (size_t)(j + 32 + l15) * HD, 0, 3);

  // ---- online softmax (rows live across 16-lane halves, one per acc VGPR)
  float p0[8], p1[8], alpha[8];
#pragma unroll
  for (int r = 0; r < 8; ++r) {
    float v0 = s0[r], v1 = s1[r];
    if (MASK) {
      const int qg = q0 + half * 8 + r;
      v0 = (j + l15      <= qg) ? v0 : -1e30f;
      v1 = (j + 16 + l15 <= qg) ? v1 : -1e30f;
    }
    float t = fmaxf(v0, v1);
#pragma unroll
    for (int off = 1; off < 16; off <<= 1)
      t = fmaxf(t, __shfl_xor(t, off, 32));
    const float mn = fmaxf(mrow[r], t);
    alpha[r] = __expf(mrow[r] - mn);
    mrow[r] = mn;
    p0[r] = __expf(v0 - mn);
    p1[r] = __expf(v1 - mn);
    float rs = p0[r] + p1[r];
#pragma unroll
    for (int off = 1; off < 16; off <<= 1)
      rs += __shfl_xor(rs, off, 32);
    lrow[r] = lrow[r] * alpha[r] + rs;
  }
#pragma unroll
  for (int t = 0; t < 4; ++t)
#pragma unroll
    for (int r = 0; r < 8; ++r)
      acc[t][r] *= alpha[r];

  // ---- transpose P (C-layout -> A-layout) through wave-private LDS
#pragma unroll
  for (int r = 0; r < 8; ++r) {
    const int m = half * 8 + r;
    ldsp[m * 32 + l15]      = (_Float16)p0[r];
    ldsp[m * 32 + 16 + l15] = (_Float16)p1[r];
  }
  asm volatile("s_wait_dscnt 0" ::: "memory");  // intra-wave LDS RAW
  v8h plo = *(const v8h*)(&ldsp[l15 * 32 + half * 16]);
  v8h phi = *(const v8h*)(&ldsp[l15 * 32 + half * 16 + 8]);
  v16h pa = cat8(plo, phi);

  // ---- out += P (16x32) * V (32x64); V pre-transposed -> contiguous loads
#pragma unroll
  for (int t = 0; t < 4; ++t) {
    v16h vb = load_b_frag(Vp + (size_t)(t * 16 + l15) * SEQ, j, half);
    acc[t] = WMMA_F16(pa, vb, acc[t]);
  }
}

// One wave owns a 16-row query tile; full unmasked tiles, then one masked
// tile straddling the diagonal.  Q,K [B,H,S,hd]; V transposed [B,H,hd,S].
__global__ __launch_bounds__(128) void attn_k(const _Float16* __restrict__ Q,
                                              const _Float16* __restrict__ K,
                                              const _Float16* __restrict__ Vt,
                                              _Float16* __restrict__ O) {
  alignas(32) __shared__ _Float16 lds_p[4][16 * 32];  // per-wave P tile
  const int lane = threadIdx.x & 31;
  const int wv   = threadIdx.x >> 5;   // 0..3
  const int l15  = lane & 15;
  const int half = lane >> 4;
  const int q0 = blockIdx.x * 64 + wv * 16;
  const int bh = blockIdx.y;           // b*16 + h

  const _Float16* Qp = Q  + (size_t)bh * SEQ * HD;
  const _Float16* Kp = K  + (size_t)bh * SEQ * HD;
  const _Float16* Vp = Vt + (size_t)bh * HD * SEQ;
  _Float16* ldsp = &lds_p[wv][0];

  // q fragments (16x64 -> two 16x32 A-frags), scale 1/sqrt(64) folded in.
  v16h qa0 = load_a_frag(Qp + (size_t)(q0 + l15) * HD, 0,  half);
  v16h qa1 = load_a_frag(Qp + (size_t)(q0 + l15) * HD, 32, half);
#pragma unroll
  for (int i = 0; i < 16; ++i) {
    qa0[i] *= (_Float16)0.125f;
    qa1[i] *= (_Float16)0.125f;
  }

  float mrow[8], lrow[8];
  v8f acc[4] = {};
#pragma unroll
  for (int r = 0; r < 8; ++r) { mrow[r] = -1e30f; lrow[r] = 0.0f; }

  const int jfull = q0 & ~31;   // keys [0, jfull) never touch the diagonal
  for (int j = 0; j < jfull; j += 32)
    attn_step<false>(Kp, Vp, ldsp, qa0, qa1, acc, mrow, lrow, j, q0, l15, half);
  attn_step<true>(Kp, Vp, ldsp, qa0, qa1, acc, mrow, lrow, jfull, q0, l15, half);

  // ---- normalize and store f16 [B,S,D]
  const int b = bh >> 4, h = bh & 15;
#pragma unroll
  for (int r = 0; r < 8; ++r) {
    const float inv = 1.0f / lrow[r];
    const int sg = q0 + half * 8 + r;
    const size_t base = ((size_t)(b * SEQ + sg)) * DM + h * HD;
#pragma unroll
    for (int t = 0; t < 4; ++t)
      O[base + t * 16 + l15] = (_Float16)(acc[t][r] * inv);
  }
}

// ----------------------------------------------------------------- launch ---
extern "C" void kernel_launch(void* const* d_in, const int* in_sizes, int n_in,
                              void* d_out, int out_size, void* d_ws, size_t ws_size,
                              hipStream_t stream) {
  const float* x  = (const float*)d_in[0];
  const float* wq = (const float*)d_in[1];
  const float* wk = (const float*)d_in[2];
  const float* wv = (const float*)d_in[3];
  const float* wo = (const float*)d_in[4];

  char* ws = (char*)d_ws;
  _Float16* xh  = (_Float16*)(ws);                       // 8 MB
  _Float16* wqh = (_Float16*)(ws + (size_t)( 8 << 20));  // 2 MB each
  _Float16* wkh = (_Float16*)(ws + (size_t)(10 << 20));
  _Float16* wvh = (_Float16*)(ws + (size_t)(12 << 20));
  _Float16* woh = (_Float16*)(ws + (size_t)(14 << 20));
  _Float16* qh  = (_Float16*)(ws + (size_t)(16 << 20));  // 8 MB
  _Float16* kh  = (_Float16*)(ws + (size_t)(24 << 20));  // 8 MB
  _Float16* vth = (_Float16*)(ws + (size_t)(32 << 20));  // 8 MB (transposed)
  _Float16* ah  = (_Float16*)(ws + (size_t)(40 << 20));  // 8 MB

  const int nx = MROWS * DM;   // 4'194'304
  const int nw = DM * DM;      // 1'048'576
  f32_to_f16_k<<<nx / 256, 256, 0, stream>>>(x,  xh,  nx);
  f32_to_f16_k<<<nw / 256, 256, 0, stream>>>(wq, wqh, nw);
  f32_to_f16_k<<<nw / 256, 256, 0, stream>>>(wk, wkh, nw);
  f32_to_f16_k<<<nw / 256, 256, 0, stream>>>(wv, wvh, nw);
  f32_to_f16_k<<<nw / 256, 256, 0, stream>>>(wo, woh, nw);

  dim3 g(MROWS / 128, DM / 64);   // 32 x 16
  gemm_wmma_k<0><<<g, 256, 0, stream>>>(xh, wqh, qh);
  gemm_wmma_k<0><<<g, 256, 0, stream>>>(xh, wkh, kh);
  gemm_wmma_k<1><<<g, 256, 0, stream>>>(xh, wvh, vth);

  rope_k<<<(BQ * NH * SEQ * 32) / 256, 256, 0, stream>>>(qh, kh);

  dim3 ga(SEQ / 64, BQ * NH);     // 32 x 32
  attn_k<<<ga, 128, 0, stream>>>(qh, kh, vth, ah);

  gemm_wmma_k<2><<<g, 256, 0, stream>>>(ah, woh, d_out);
}
